// MambaBlock_10823317586393
// MI455X (gfx1250) — matrix operand order repaired and checked by reading, not compiled, not run on hardware
//
#include <hip/hip_runtime.h>
#include <hip/hip_bf16.h>

// ---------------------------------------------------------------------------
// Mamba block for MI455X (gfx1250, wave32, WMMA).
// All five GEMMs use v_wmma_f32_16x16x32_bf16 (f32 accumulate) with fused
// epilogues, NB-wide N register blocking, ping-pong double-buffered K loop,
// and single-32B-load fragments (A stored K-swizzled so each lane's fragment
// is 16 contiguous bf16).  Scan is 1 block per batch, 512 lanes, state in
// registers, with register double-buffering of all per-step inputs so global
// load latency is off the 512-step sequential critical path.
// ---------------------------------------------------------------------------

#define B_  32
#define L_  512
#define D_  256
#define DIN 512
#define BL  (B_ * L_)   // 16384

typedef __bf16 bf16_t;
typedef __attribute__((ext_vector_type(16))) bf16_t bf16x16;
typedef __attribute__((ext_vector_type(8)))  float  v8f;

__device__ __forceinline__ unsigned short f2bf(float f) {
    unsigned int x = __float_as_uint(f);
    unsigned int r = x + 0x7FFFu + ((x >> 16) & 1u);   // round-to-nearest-even
    return (unsigned short)(r >> 16);
}

__device__ __forceinline__ float silu_f(float v) {
    return v / (1.f + __expf(-v));
}

// A-operand store swizzle: within each 32-element K block, swap the middle two
// 8-element chunks (2-bit reverse of chunk id): chunk 0->0, 1->2, 2->1, 3->3.
// Consumer lane group g then reads 16 contiguous elements at k0 + 16*g which
// are exactly K [8g..8g+7] ++ [8g+16..8g+23] in WMMA element order.
__device__ __forceinline__ int aswz(int k) {
    int c = (k >> 3) & 3;
    c = ((c & 1) << 1) | (c >> 1);
    return (k & ~31) | (c << 3) | (k & 7);
}

// single 32-byte fragment load (16 contiguous bf16, 32B aligned)
__device__ __forceinline__ bf16x16 loadFrag(const unsigned short* p, int k0) {
    return *(const bf16x16*)(p + k0);
}
__device__ __forceinline__ bf16x16 loadFragB(const unsigned short* p, int k0) {
    typedef __attribute__((ext_vector_type(8))) bf16_t bf16x8;
    bf16x8 lo = *(const bf16x8*)(p + k0);
    bf16x8 hi = *(const bf16x8*)(p + k0 + 16);
    return __builtin_shufflevector(lo, hi, 0,1,2,3,4,5,6,7,8,9,10,11,12,13,14,15);
}

// ---------------------------------------------------------------------------
// mask: hid0 = hidden * valid_mask   (also the residual for stage 1)
// ---------------------------------------------------------------------------
__global__ __launch_bounds__(256)
void mask_kernel(const float* __restrict__ hid, const int* __restrict__ msk,
                 float* __restrict__ out) {
    int i = blockIdx.x * 256 + threadIdx.x;     // BL*D_ elements
    if (i < BL * D_) out[i] = hid[i] * (float)msk[i >> 8];   // D_=256
}

// ---------------------------------------------------------------------------
// LayerNorm over D_=256, one block per row, bf16 output (A-swizzled columns).
// ---------------------------------------------------------------------------
__global__ __launch_bounds__(256)
void ln_bf16_kernel(const float* __restrict__ x, const float* __restrict__ g,
                    const float* __restrict__ beta, unsigned short* __restrict__ out) {
    const int row = blockIdx.x, t = threadIdx.x;
    float v = x[(size_t)row * D_ + t];
    __shared__ float sm[8];
    float s = v;
    #pragma unroll
    for (int o = 16; o > 0; o >>= 1) s += __shfl_down(s, o, 32);
    if ((t & 31) == 0) sm[t >> 5] = s;
    __syncthreads();
    float mu = 0.f;
    #pragma unroll
    for (int i = 0; i < 8; ++i) mu += sm[i];
    mu *= (1.f / 256.f);
    __syncthreads();
    float d = v - mu;
    float q = d * d;
    #pragma unroll
    for (int o = 16; o > 0; o >>= 1) q += __shfl_down(q, o, 32);
    if ((t & 31) == 0) sm[t >> 5] = q;
    __syncthreads();
    float var = 0.f;
    #pragma unroll
    for (int i = 0; i < 8; ++i) var += sm[i];
    var *= (1.f / 256.f);
    float y = d * rsqrtf(var + 1e-5f) * g[t] + beta[t];
    out[(size_t)row * D_ + aswz(t)] = f2bf(y);
}

// ---------------------------------------------------------------------------
// generic f32 -> bf16 conversion (weights, consumed as B: linear layout)
// ---------------------------------------------------------------------------
__global__ __launch_bounds__(256)
void cvt_bf16_kernel(const float* __restrict__ in, unsigned short* __restrict__ out, int n) {
    int i = blockIdx.x * 256 + threadIdx.x;
    if (i < n) out[i] = f2bf(in[i]);
}

// ---------------------------------------------------------------------------
// WMMA GEMM:  C[M,N] = A[M,K] * W[N,K]^T     (bf16 operands, f32 accumulate)
// A is stored K-swizzled (aswz), W is linear.  One wave computes a
// 16 x (16*NB) strip; K loop unrolled 2x with ping-pong fragment sets.
// Requires K % 64 == 0 (true for K=256/512/1024).
// EPI: 0=store f32, 1=+resid f32, 2=+bias,SiLU -> swizzled bf16,
//      3=+bias,+resid,*mask -> f32
// ---------------------------------------------------------------------------
template <int EPI, int NB>
__global__ __launch_bounds__(256)
void wmma_gemm_bf16(const unsigned short* __restrict__ A,
                    const unsigned short* __restrict__ W,
                    int M, int N, int K,
                    float* __restrict__ outF,
                    unsigned short* __restrict__ outBf,
                    const float* __restrict__ bias,
                    const float* __restrict__ resid,
                    const int* __restrict__ msk) {
    const int lane = threadIdx.x & 31;
    const int wave = threadIdx.x >> 5;
    const int tilesN  = N >> 4;
    const int groupsN = tilesN / NB;
    const int wt = blockIdx.x * 8 + wave;
    if (wt >= (M >> 4) * groupsN) return;       // wave-uniform: EXEC stays all-1s
    const int tm = wt / groupsN;
    const int tg = wt - tm * groupsN;           // N-group index
    const int grp = lane >> 4;
    const int mrow = tm * 16 + (lane & 15);

    const unsigned short* arow = A + (size_t)mrow * K + grp * 16;
    const unsigned short* brow[NB];
    #pragma unroll
    for (int j = 0; j < NB; ++j)
        brow[j] = W + (size_t)((tg * NB + j) * 16 + (lane & 15)) * K + grp * 8;

    v8f acc[NB];
    #pragma unroll
    for (int j = 0; j < NB; ++j) acc[j] = (v8f){0.f,0.f,0.f,0.f,0.f,0.f,0.f,0.f};

    // prologue: fill both pipeline stages (K-steps 0 and 32)
    bf16x16 aA = loadFrag(arow, 0);
    bf16x16 bA[NB];
    #pragma unroll
    for (int j = 0; j < NB; ++j) bA[j] = loadFragB(brow[j], 0);
    bf16x16 aB = loadFrag(arow, 32);
    bf16x16 bB[NB];
    #pragma unroll
    for (int j = 0; j < NB; ++j) bB[j] = loadFragB(brow[j], 32);

    #pragma unroll 1
    for (int k = 64; k < K; k += 64) {
        // consume set A (set B loads in flight), then refill set A from step k
        #pragma unroll
        for (int j = 0; j < NB; ++j)
            acc[j] = __builtin_amdgcn_wmma_f32_16x16x32_bf16(
                         false, aA, false, bA[j], (short)0, acc[j], false, false);
        if (k + 64 < K) {
            __builtin_prefetch(arow + k + 64, 0, 1);
            #pragma unroll
            for (int j = 0; j < NB; ++j) __builtin_prefetch(brow[j] + k + 64, 0, 1);
        }
        aA = loadFrag(arow, k);
        #pragma unroll
        for (int j = 0; j < NB; ++j) bA[j] = loadFragB(brow[j], k);

        // consume set B (set A loads in flight), then refill set B from k+32
        #pragma unroll
        for (int j = 0; j < NB; ++j)
            acc[j] = __builtin_amdgcn_wmma_f32_16x16x32_bf16(
                         false, aB, false, bB[j], (short)0, acc[j], false, false);
        aB = loadFrag(arow, k + 32);
        #pragma unroll
        for (int j = 0; j < NB; ++j) bB[j] = loadFragB(brow[j], k + 32);
    }
    // drain both stages
    #pragma unroll
    for (int j = 0; j < NB; ++j)
        acc[j] = __builtin_amdgcn_wmma_f32_16x16x32_bf16(
                     false, aA, false, bA[j], (short)0, acc[j], false, false);
    #pragma unroll
    for (int j = 0; j < NB; ++j)
        acc[j] = __builtin_amdgcn_wmma_f32_16x16x32_bf16(
                     false, aB, false, bB[j], (short)0, acc[j], false, false);

    // epilogue: lane l, reg r -> row tm*16 + 8*(l>>4) + r, col base + (l&15)
    #pragma unroll
    for (int j = 0; j < NB; ++j) {
        int nc = (tg * NB + j) * 16 + (lane & 15);
        #pragma unroll
        for (int r = 0; r < 8; ++r) {
            int mr = tm * 16 + grp * 8 + r;
            size_t idx = (size_t)mr * N + nc;
            float v = acc[j][r];
            if (EPI == 2) {
                v += bias[nc];
                v = silu_f(v);
                // output consumed as A of the next GEMM: swizzled store
                outBf[(size_t)mr * N + aswz(nc)] = f2bf(v);
            } else if (EPI == 3) {
                v += bias[nc] + resid[idx];
                v *= (float)msk[mr];          // mr == flat (b*L + l)
                outF[idx] = v;
            } else {
                if (EPI == 1) v += resid[idx];
                outF[idx] = v;
            }
        }
    }
}

// ---------------------------------------------------------------------------
// depthwise causal conv (D_CONV=4) + SiLU over x half of xz.
// Emits f32 (linear, for scan) + bf16 (A-swizzled, for x_proj GEMM).
// ---------------------------------------------------------------------------
__global__ __launch_bounds__(256)
void conv_silu_kernel(const float* __restrict__ xz, const float* __restrict__ cw,
                      const float* __restrict__ cb, float* __restrict__ xact,
                      unsigned short* __restrict__ xbf) {
    int idx = blockIdx.x * 256 + threadIdx.x;      // BL * DIN
    if (idx >= BL * DIN) return;
    int c  = idx & (DIN - 1);
    int bl = idx >> 9;
    int l  = bl & (L_ - 1);
    int b  = bl >> 9;
    float acc = cb[c];
    #pragma unroll
    for (int k = 0; k < 4; ++k) {
        int ls = l + k - 3;
        if (ls >= 0)
            acc += xz[((size_t)(b * L_ + ls)) * (2 * DIN) + c] * cw[c * 4 + k];
    }
    float s = silu_f(acc);
    xact[idx] = s;
    xbf[(size_t)bl * DIN + aswz(c)] = f2bf(s);
}

// ---------------------------------------------------------------------------
// dt = softplus(dbc[:, :16] @ dt_proj_w^T + dt_proj_b)   (K=16: scalar MACs)
// ---------------------------------------------------------------------------
__global__ __launch_bounds__(256)
void dt_kernel(const float* __restrict__ dbc, const float* __restrict__ dtw,
               const float* __restrict__ dtb, float* __restrict__ dt) {
    int idx = blockIdx.x * 256 + threadIdx.x;      // BL * DIN
    if (idx >= BL * DIN) return;
    int d  = idx & (DIN - 1);
    int bl = idx >> 9;
    const float* dr = dbc + (size_t)bl * 48;
    float a = dtb[d];
    #pragma unroll
    for (int r = 0; r < 16; ++r) a += dr[r] * dtw[d * 16 + r];
    dt[idx] = (a > 20.f) ? a : log1pf(__expf(a));
}

// ---------------------------------------------------------------------------
// selective scan: 1 block per batch, lane d holds h[16] in registers.
// All per-step inputs (B/C row, u, dt, z) are double-buffered in registers:
// step l+1's global loads issue right after the barrier and complete while
// step l's 16-deep exp/FMA chain executes, keeping global latency off the
// 512-step sequential critical path.
// Fused: y = (scan + u*D) * silu(z), emitted bf16 A-swizzled for out_proj.
// ---------------------------------------------------------------------------
__global__ __launch_bounds__(512)
void scan_kernel(const float* __restrict__ xact, const float* __restrict__ dt,
                 const float* __restrict__ dbc, const float* __restrict__ xz,
                 const float* __restrict__ Alog, const float* __restrict__ Dp,
                 unsigned short* __restrict__ ybf) {
    const int b = blockIdx.x;
    const int d = threadIdx.x;                 // 0..511
    const int dsw = aswz(d);
    __shared__ float sB[16], sC[16];
    float A[16], h[16];
    #pragma unroll
    for (int n = 0; n < 16; ++n) {
        A[n] = -__expf(Alog[d * 16 + n]);
        h[n] = 0.f;
    }
    const float Dv = Dp[d];
    const size_t base = (size_t)b * L_;

    // preload step 0 into the register pipeline
    float bcN = (d < 32) ? dbc[base * 48 + 16 + d] : 0.f;
    float uN  = xact[base * DIN + d];
    float dtN = dt[base * DIN + d];
    float zN  = xz[base * (2 * DIN) + DIN + d];

    for (int l = 0; l < L_; ++l) {
        __syncthreads();                       // previous step done reading sB/sC
        if (d < 32) {
            if (d < 16) sB[d] = bcN; else sC[d - 16] = bcN;
        }
        float u = uN, dtv = dtN, z = zN;
        __syncthreads();

        // issue step l+1 loads now; they complete during the compute chain
        if (l + 1 < L_) {
            size_t bl1 = base + l + 1;
            if (d < 32) bcN = dbc[bl1 * 48 + 16 + d];
            uN  = xact[bl1 * DIN + d];
            dtN = dt[bl1 * DIN + d];
            zN  = xz[bl1 * (2 * DIN) + DIN + d];
        }

        float du = dtv * u;
        float y = 0.f;
        #pragma unroll
        for (int n = 0; n < 16; ++n) {
            float dA = __expf(dtv * A[n]);
            h[n] = dA * h[n] + du * sB[n];
            y += h[n] * sC[n];
        }
        y += u * Dv;
        y *= silu_f(z);
        ybf[(base + l) * DIN + dsw] = f2bf(y);
    }
}

// ---------------------------------------------------------------------------
extern "C" void kernel_launch(void* const* d_in, const int* in_sizes, int n_in,
                              void* d_out, int out_size, void* d_ws, size_t ws_size,
                              hipStream_t stream) {
    (void)in_sizes; (void)n_in; (void)out_size; (void)ws_size;
    const float* hidden = (const float*)d_in[0];
    const int*   vmask  = (const int*)  d_in[1];
    const float* ln1g   = (const float*)d_in[2];
    const float* ln1b   = (const float*)d_in[3];
    const float* ln2g   = (const float*)d_in[4];
    const float* ln2b   = (const float*)d_in[5];
    const float* w_in   = (const float*)d_in[6];    // (1024, 256)
    const float* convw  = (const float*)d_in[7];    // (512, 4)
    const float* convb  = (const float*)d_in[8];
    const float* w_xp   = (const float*)d_in[9];    // (48, 512)
    const float* dtw    = (const float*)d_in[10];   // (512, 16)
    const float* dtb    = (const float*)d_in[11];
    const float* Alog   = (const float*)d_in[12];   // (512, 16)
    const float* Dp     = (const float*)d_in[13];
    const float* w_out  = (const float*)d_in[14];   // (256, 512)
    const float* w1     = (const float*)d_in[15];   // (1024, 256)
    const float* b1     = (const float*)d_in[16];
    const float* w2     = (const float*)d_in[17];   // (256, 1024)
    const float* b2     = (const float*)d_in[18];
    float* out = (float*)d_out;

    char* p = (char*)d_ws;
    auto take = [&](size_t bytes) -> char* {
        char* r = p; p += (bytes + 255) & ~(size_t)255; return r;
    };
    float*          hid0   = (float*)         take((size_t)BL * D_ * 4);
    unsigned short* xn_bf  = (unsigned short*)take((size_t)BL * D_ * 2);
    unsigned short* win_bf = (unsigned short*)take((size_t)1024 * 256 * 2);
    float*          xz     = (float*)         take((size_t)BL * 1024 * 4);
    float*          xact   = (float*)         take((size_t)BL * DIN * 4);
    unsigned short* x_bf   = (unsigned short*)take((size_t)BL * DIN * 2);
    unsigned short* wxp_bf = (unsigned short*)take((size_t)48 * 512 * 2);
    float*          dbc    = (float*)         take((size_t)BL * 48 * 4);
    float*          dt     = (float*)         take((size_t)BL * DIN * 4);
    unsigned short* y_bf   = (unsigned short*)take((size_t)BL * DIN * 2);
    unsigned short* wo_bf  = (unsigned short*)take((size_t)256 * 512 * 2);
    float*          hid2   = (float*)         take((size_t)BL * D_ * 4);
    unsigned short* x2_bf  = (unsigned short*)take((size_t)BL * D_ * 2);
    unsigned short* w1_bf  = (unsigned short*)take((size_t)1024 * 256 * 2);
    unsigned short* m1_bf  = (unsigned short*)take((size_t)BL * 1024 * 2);
    unsigned short* w2_bf  = (unsigned short*)take((size_t)256 * 1024 * 2);

    auto gemm_blocks = [](int M, int N, int NB) {
        return ((M >> 4) * ((N >> 4) / NB) + 7) / 8;
    };

    // 1. mask + residual snapshot
    mask_kernel<<<(BL * D_) / 256, 256, 0, stream>>>(hidden, vmask, hid0);
    // 2. LN1 -> bf16 (A-swizzled)
    ln_bf16_kernel<<<BL, 256, 0, stream>>>(hid0, ln1g, ln1b, xn_bf);
    // 3. weight conversions (B operands: linear)
    cvt_bf16_kernel<<<(1024 * 256 + 255) / 256, 256, 0, stream>>>(w_in, win_bf, 1024 * 256);
    cvt_bf16_kernel<<<(48 * 512 + 255) / 256, 256, 0, stream>>>(w_xp, wxp_bf, 48 * 512);
    cvt_bf16_kernel<<<(256 * 512 + 255) / 256, 256, 0, stream>>>(w_out, wo_bf, 256 * 512);
    cvt_bf16_kernel<<<(1024 * 256 + 255) / 256, 256, 0, stream>>>(w1, w1_bf, 1024 * 256);
    cvt_bf16_kernel<<<(256 * 1024 + 255) / 256, 256, 0, stream>>>(w2, w2_bf, 256 * 1024);
    // 4. in_proj: xz = xn @ w_in^T           (M=16384, N=1024, K=256)
    wmma_gemm_bf16<0, 4><<<gemm_blocks(BL, 1024, 4), 256, 0, stream>>>(
        xn_bf, win_bf, BL, 1024, 256, xz, nullptr, nullptr, nullptr, nullptr);
    // 5. depthwise causal conv + SiLU
    conv_silu_kernel<<<(BL * DIN + 255) / 256, 256, 0, stream>>>(xz, convw, convb, xact, x_bf);
    // 6. x_proj: dbc = x @ w_xp^T            (M=16384, N=48, K=512)
    wmma_gemm_bf16<0, 3><<<gemm_blocks(BL, 48, 3), 256, 0, stream>>>(
        x_bf, wxp_bf, BL, 48, 512, dbc, nullptr, nullptr, nullptr, nullptr);
    // 7. dt = softplus(dbc[:, :16] @ dtw^T + dtb)
    dt_kernel<<<(BL * DIN + 255) / 256, 256, 0, stream>>>(dbc, dtw, dtb, dt);
    // 8. selective scan (+ D skip + SiLU(z) gate) -> y bf16 (A-swizzled)
    scan_kernel<<<B_, DIN, 0, stream>>>(xact, dt, dbc, xz, Alog, Dp, y_bf);
    // 9. out_proj + residual: hid2 = y @ w_out^T + hid0   (N=256, K=512)
    wmma_gemm_bf16<1, 4><<<gemm_blocks(BL, 256, 4), 256, 0, stream>>>(
        y_bf, wo_bf, BL, 256, 512, hid2, nullptr, nullptr, hid0, nullptr);
    // 10. LN2 -> bf16 (A-swizzled)
    ln_bf16_kernel<<<BL, 256, 0, stream>>>(hid2, ln2g, ln2b, x2_bf);
    // 11. FFN1: m1 = SiLU(x2 @ w1^T + b1) -> bf16 (A-swizzled)   (N=1024, K=256)
    wmma_gemm_bf16<2, 4><<<gemm_blocks(BL, 1024, 4), 256, 0, stream>>>(
        x2_bf, w1_bf, BL, 1024, 256, nullptr, m1_bf, b1, nullptr, nullptr);
    // 12. FFN2 + residual + mask -> out      (N=256, K=1024)
    wmma_gemm_bf16<3, 4><<<gemm_blocks(BL, 256, 4), 256, 0, stream>>>(
        m1_bf, w2_bf, BL, 256, 1024, out, nullptr, b2, hid2, vmask);
}
